// Self_Attention_24026047054692
// MI455X (gfx1250) — matrix operand order repaired and checked by reading, not compiled, not run on hardware
//
#include <hip/hip_runtime.h>
#include <hip/hip_bf16.h>
#include <math.h>

// ---------------------------------------------------------------------------
// Self-attention (B=4, N=2048, DIM=1024, H=16, HD=64) for gfx1250 (MI455X).
// f16 WMMA (v_wmma_f32_16x16x32_f16) with f32 accumulation.
// All WMMA operand fetches are contiguous 2 x ds_load_b128 per lane.
// Tile staging uses GLOBAL_LOAD_ASYNC_TO_LDS_B128 (ASYNCcnt) when the
// toolchain exposes it, with a synchronous fallback otherwise.
// Flash-style fused attention keeps the 1.07 GB score tensor in LDS.
// ---------------------------------------------------------------------------

typedef _Float16 half_t;
typedef half_t v16h __attribute__((ext_vector_type(16)));
typedef half_t v8h  __attribute__((ext_vector_type(8)));
typedef float  v8f  __attribute__((ext_vector_type(8)));
typedef int    v4i  __attribute__((ext_vector_type(4)));

#define WMMA16(a, b, c) \
  __builtin_amdgcn_wmma_f32_16x16x32_f16(false, (a), false, (b), (short)0, (c), false, false)

#if defined(__has_builtin)
#if __has_builtin(__builtin_amdgcn_global_load_async_to_lds_b128) && \
    __has_builtin(__builtin_amdgcn_s_wait_asynccnt)
#define HAS_ASYNC_LDS 1
#endif
#endif
#ifndef HAS_ASYNC_LDS
#define HAS_ASYNC_LDS 0
#endif

// problem constants
constexpr int BATCH = 4;
constexpr int NSEQ  = 2048;
constexpr int DIMSZ = 1024;
constexpr int HEADS = 16;
constexpr int INTER = 1024;
constexpr int HD    = 64;
constexpr float SCALE = 0.03125f;   // INTER^-0.5 = 1/32

// ---------------------------------------------------------------------------
// async staging helpers
// ---------------------------------------------------------------------------
__device__ __forceinline__ void async_wait0() {
#if HAS_ASYNC_LDS
  __builtin_amdgcn_s_wait_asynccnt(0);
#endif
}

// move 16 halfs (32B) per lane from global to LDS
__device__ __forceinline__ void stage16(const half_t* __restrict__ g,
                                        half_t* __restrict__ s) {
#if HAS_ASYNC_LDS
  typedef __attribute__((address_space(1))) v4i* gp_t;   // global int4*
  typedef __attribute__((address_space(3))) v4i* lp_t;   // LDS int4*
  __builtin_amdgcn_global_load_async_to_lds_b128(
      (gp_t)(unsigned long long)g,
      (lp_t)(unsigned int)(unsigned long long)s, 0, 0);
  __builtin_amdgcn_global_load_async_to_lds_b128(
      (gp_t)(unsigned long long)(g + 8),
      (lp_t)(unsigned int)(unsigned long long)(s + 8), 0, 0);
#else
  *(v8h*)s       = *(const v8h*)g;
  *(v8h*)(s + 8) = *(const v8h*)(g + 8);
#endif
}

// ---------------------------------------------------------------------------
// WMMA fragment loads (CDNA5 ISA 7.12.2 layouts, wave32). Every load is a
// pair of 16B LDS transactions combined with a shuffle.
// ---------------------------------------------------------------------------
__device__ __forceinline__ v16h combine16(v8h lo, v8h hi) {
  return __builtin_shufflevector(lo, hi, 0, 1, 2, 3, 4, 5, 6, 7,
                                         8, 9, 10, 11, 12, 13, 14, 15);
}

// A fragment: 16x32 f16 tile, row-major, leading dim `ld` (multiple of 8).
__device__ __forceinline__ v16h load_a_frag(const half_t* __restrict__ s, int ld) {
  const int lane = threadIdx.x & 31;
  const half_t* p = s + (lane & 15) * ld + (((lane >> 4) & 1) << 3);
  v8h lo = *(const v8h*)(p);
  v8h hi = *(const v8h*)(p + 16);
  return combine16(lo, hi);
}

// B fragment from a K-contiguous (transposed) tile: memory holds T[col][k].
__device__ __forceinline__ v16h load_bt_frag(const half_t* __restrict__ s, int ld) {
  const int lane = threadIdx.x & 31;
  const half_t* p = s + (lane & 15) * ld + (((lane >> 4) & 1) << 3);
  v8h lo = *(const v8h*)(p);
  v8h hi = *(const v8h*)(p + 16);
  return combine16(lo, hi);
}

// ---------------------------------------------------------------------------
// fp32 -> fp16 down-conversion (grid-stride)
// ---------------------------------------------------------------------------
__global__ void f32_to_f16(const float* __restrict__ src, half_t* __restrict__ dst, int n) {
  int i = blockIdx.x * blockDim.x + threadIdx.x;
  const int stride = gridDim.x * blockDim.x;
  for (; i < n; i += stride) dst[i] = (half_t)src[i];
}

// fp32 [R][C] row-major  ->  fp16 [C][R] (transpose), 32x32 LDS tiles.
__global__ void __launch_bounds__(256)
f32_to_f16_T(const float* __restrict__ src, half_t* __restrict__ dst, int R, int C) {
  __shared__ float tile[32][33];
  const int c0 = blockIdx.x * 32, r0 = blockIdx.y * 32;
  const int tx = threadIdx.x & 31, ty = threadIdx.x >> 5;   // 32 x 8
#pragma unroll
  for (int i = 0; i < 32; i += 8)
    tile[ty + i][tx] = src[(size_t)(r0 + ty + i) * C + c0 + tx];
  __syncthreads();
#pragma unroll
  for (int i = 0; i < 32; i += 8)
    dst[(size_t)(c0 + ty + i) * R + r0 + tx] = (half_t)tile[tx][ty + i];
}

// ---------------------------------------------------------------------------
// Tiled WMMA GEMM: C[M,N] = A[M,K] * B[K,N] with B supplied TRANSPOSED
// (BT layout [N][K]). BM=128, BN=128, BK=32; 256 thr = 8 waves.
// Double-buffered LDS staging: async loads of tile t+1 overlap WMMAs on t.
// MODE 0: store f32 C.
// MODE 1: scatter f16 into Q[B][H][N][HD], K[B][H][N][HD], V[B][H][HD][N].
// ---------------------------------------------------------------------------
constexpr int BM = 128, BN = 128, BK = 32;
constexpr int LDA_S = BK + 8;    // 40 halfs, rows 16B-aligned
constexpr int LDB_S = BK + 8;

template <int MODE>
__global__ void __launch_bounds__(256)
gemm_f16(const half_t* __restrict__ A, const half_t* __restrict__ BT,
         float* __restrict__ C,
         half_t* __restrict__ Qo, half_t* __restrict__ Ko, half_t* __restrict__ Vo,
         int M, int Nc, int Kc) {
  __shared__ alignas(16) half_t As[2][BM * LDA_S];   // 2 x 10240 B
  __shared__ alignas(16) half_t Bs[2][BN * LDB_S];   // 2 x 10240 B

  const int tid  = threadIdx.x;
  const int lane = tid & 31;
  const int wid  = tid >> 5;
  const int wm   = wid & 3;     // 4 waves along M (32 rows each)
  const int wn   = wid >> 2;    // 2 waves along N (64 cols each)

  const int rowBase = blockIdx.y * BM;
  const int colBase = blockIdx.x * BN;

  v8f acc[2][4] = {};

  // staging coordinates: 128 rows x 2 segments of 16 halfs
  const int sr = tid >> 1, scs = (tid & 1) * 16;
  const half_t* gA = A + (size_t)(rowBase + sr) * Kc + scs;
  const half_t* gB = BT + (size_t)(colBase + sr) * Kc + scs;
  half_t* sA = &As[0][sr * LDA_S + scs];
  half_t* sB = &Bs[0][sr * LDB_S + scs];

  // prologue: stage tile 0 into buffer 0
  stage16(gA, sA);
  stage16(gB, sB);

  const int T = Kc / BK;
  for (int t = 0; t < T; ++t) {
    async_wait0();        // tile t resident in LDS (this wave's portion)
    __syncthreads();      // all waves' portions visible; buf (t+1)&1 free
    if (t + 1 < T) {
      const int nb = (t + 1) & 1;
      stage16(gA + (size_t)(t + 1) * BK, sA + nb * (BM * LDA_S));
      stage16(gB + (size_t)(t + 1) * BK, sB + nb * (BN * LDB_S));
    }
    const half_t* ab = &As[t & 1][0];
    const half_t* bb = &Bs[t & 1][0];
    v16h a0 = load_a_frag(ab + (wm * 32) * LDA_S, LDA_S);
    v16h a1 = load_a_frag(ab + (wm * 32 + 16) * LDA_S, LDA_S);
#pragma unroll
    for (int j = 0; j < 4; ++j) {
      v16h bf = load_bt_frag(bb + (wn * 64 + j * 16) * LDB_S, LDB_S);
      acc[0][j] = WMMA16(a0, bf, acc[0][j]);
      acc[1][j] = WMMA16(a1, bf, acc[1][j]);
    }
  }

  // epilogue
  const int cr = ((lane >> 4) & 1) * 8;
  const int cc = lane & 15;
#pragma unroll
  for (int i = 0; i < 2; ++i) {
#pragma unroll
    for (int j = 0; j < 4; ++j) {
#pragma unroll
      for (int r = 0; r < 8; ++r) {
        const int gr = rowBase + wm * 32 + i * 16 + cr + r;
        const int gc = colBase + wn * 64 + j * 16 + cc;
        const float v = acc[i][j][r];
        if (MODE == 0) {
          C[(size_t)gr * Nc + gc] = v;
        } else {
          // gr = b*NSEQ + n ; gc = chunk*INTER + h*HD + d
          const int b = gr >> 11, n = gr & (NSEQ - 1);
          const int chunk = gc >> 10;
          const int w = gc & (INTER - 1);
          const int h = w >> 6, d = w & (HD - 1);
          if (chunk == 0) {
            Qo[(((size_t)(b * HEADS + h)) * NSEQ + n) * HD + d] = (half_t)v;
          } else if (chunk == 1) {
            Ko[(((size_t)(b * HEADS + h)) * NSEQ + n) * HD + d] = (half_t)v;
          } else {  // V stored head-transposed: [B][H][HD][N]
            Vo[(((size_t)(b * HEADS + h)) * HD + d) * NSEQ + n] = (half_t)v;
          }
        }
      }
    }
  }
}

// ---------------------------------------------------------------------------
// Flash attention: one block per (b*h, 64-row Q tile); 256 threads = 8 waves.
// Each wave owns a 16x32 slice of the 64x64 S / O tiles (2 frags).
// Q,K: [B][H][N][HD]  (K-contiguous for both A and B^T fetches)
// V:   [B][H][HD][N]  (n-contiguous -> B^T fetch for P.V)
// ---------------------------------------------------------------------------
constexpr int BR = 64, BC = 64;
constexpr int LQ = HD + 8;    // 72 halfs
constexpr int LV = BC + 8;    // 72 halfs (Vt is [d][n])
constexpr int LS = BC + 4;    // 68 floats, rows 16B-aligned for float4
constexpr int LP = BC + 8;    // 72 halfs

__global__ void __launch_bounds__(256)
flash_attn(const half_t* __restrict__ Q, const half_t* __restrict__ K,
           const half_t* __restrict__ V, half_t* __restrict__ O) {
  __shared__ alignas(16) half_t Qs[BR * LQ];
  __shared__ alignas(16) half_t Ks[BC * LQ];
  __shared__ alignas(16) half_t Vt[HD * LV];
  __shared__ alignas(16) float  Sb[BR * LS];
  __shared__ alignas(16) half_t Pb[BR * LP];
  __shared__ float rmax[BR], rsum[BR], rscale[BR];
  __shared__ float red[BR][4];

  const int tid  = threadIdx.x;
  const int lane = tid & 31;
  const int wid  = tid >> 5;
  const int wm   = wid & 3;      // 16-row strip of the 64-row tile
  const int wn   = wid >> 2;     // 32-col strip
  const int cr   = ((lane >> 4) & 1) * 8;
  const int cc   = lane & 15;

  const int bh = blockIdx.x;
  const int b  = bh >> 4, h = bh & 15;
  const int qt = blockIdx.y;

  const half_t* Qg = Q + (((size_t)(b * HEADS + h)) * NSEQ + (size_t)qt * BR) * HD;
  const half_t* Kg = K + (((size_t)(b * HEADS + h)) * NSEQ) * HD;
  const half_t* Vg = V + ((size_t)(b * HEADS + h)) * HD * NSEQ;   // [HD][NSEQ]

  const int strow = tid >> 2, stseg = (tid & 3) * 16;   // staging: 64 x 4 segs

  // stage Q tile (pre-scaled by 1/sqrt(INTER))
  {
    const v8h* g = (const v8h*)(Qg + (size_t)strow * HD + stseg);
    v8h x0 = g[0], x1 = g[1];
#pragma unroll
    for (int e = 0; e < 8; ++e) {
      x0[e] = (half_t)((float)x0[e] * SCALE);
      x1[e] = (half_t)((float)x1[e] * SCALE);
    }
    *(v8h*)&Qs[strow * LQ + stseg]     = x0;
    *(v8h*)&Qs[strow * LQ + stseg + 8] = x1;
  }
  if (tid < BR) { rmax[tid] = -1e30f; rsum[tid] = 0.0f; }

  v8f o0 = {}, o1 = {};

  for (int kt = 0; kt < NSEQ / BC; ++kt) {
    __syncthreads();                       // previous users of Ks/Vt done
    // stage K tile [n][d] and V^T tile [d][n] via async-to-LDS
    stage16(Kg + ((size_t)(kt * BC + strow)) * HD + stseg,
            &Ks[strow * LQ + stseg]);
    stage16(Vg + (size_t)strow * NSEQ + kt * BC + stseg,
            &Vt[strow * LV + stseg]);
    async_wait0();
    __syncthreads();

    // S = Q * K^T  (64x64, each wave 2 frags), K-dim = HD = 2 WMMA steps
    v8f s0 = {}, s1 = {};
#pragma unroll
    for (int k0 = 0; k0 < HD; k0 += 32) {
      v16h a  = load_a_frag(&Qs[(wm * 16) * LQ + k0], LQ);
      v16h b0 = load_bt_frag(&Ks[(wn * 32) * LQ + k0], LQ);
      v16h b1 = load_bt_frag(&Ks[(wn * 32 + 16) * LQ + k0], LQ);
      s0 = WMMA16(a, b0, s0);
      s1 = WMMA16(a, b1, s1);
    }
#pragma unroll
    for (int r = 0; r < 8; ++r) {
      Sb[(wm * 16 + cr + r) * LS + wn * 32 + cc]      = s0[r];
      Sb[(wm * 16 + cr + r) * LS + wn * 32 + 16 + cc] = s1[r];
    }
    __syncthreads();

    // online softmax: 4 threads per row, 16 cols each (float4 reads)
    {
      const float4* srow = (const float4*)&Sb[strow * LS + stseg];
      float m = -1e30f;
#pragma unroll
      for (int c4 = 0; c4 < 4; ++c4) {
        const float4 sv = srow[c4];
        m = fmaxf(m, fmaxf(fmaxf(sv.x, sv.y), fmaxf(sv.z, sv.w)));
      }
      red[strow][tid & 3] = m;
    }
    __syncthreads();
    if (tid < BR) {
      const float m = fmaxf(fmaxf(red[tid][0], red[tid][1]),
                            fmaxf(red[tid][2], red[tid][3]));
      const float mnew = fmaxf(rmax[tid], m);
      rscale[tid] = __expf(rmax[tid] - mnew);
      rmax[tid]   = mnew;
    }
    __syncthreads();
    {
      const float mnew = rmax[strow];
      const float4* srow = (const float4*)&Sb[strow * LS + stseg];
      float psum = 0.0f;
      v16h pv;
#pragma unroll
      for (int c4 = 0; c4 < 4; ++c4) {
        const float4 sv = srow[c4];
        const float p0 = __expf(sv.x - mnew);
        const float p1 = __expf(sv.y - mnew);
        const float p2 = __expf(sv.z - mnew);
        const float p3 = __expf(sv.w - mnew);
        psum += (p0 + p1) + (p2 + p3);
        pv[c4 * 4 + 0] = (half_t)p0;
        pv[c4 * 4 + 1] = (half_t)p1;
        pv[c4 * 4 + 2] = (half_t)p2;
        pv[c4 * 4 + 3] = (half_t)p3;
      }
      *(v8h*)&Pb[strow * LP + stseg] =
          __builtin_shufflevector(pv, pv, 0, 1, 2, 3, 4, 5, 6, 7);
      *(v8h*)&Pb[strow * LP + stseg + 8] =
          __builtin_shufflevector(pv, pv, 8, 9, 10, 11, 12, 13, 14, 15);
      red[strow][tid & 3] = psum;
    }
    __syncthreads();
    if (tid < BR) {
      rsum[tid] = rsum[tid] * rscale[tid] +
                  red[tid][0] + red[tid][1] + red[tid][2] + red[tid][3];
    }
    __syncthreads();

    // rescale running O, then O += P * V   (V^T tile -> contiguous B frags)
#pragma unroll
    for (int r = 0; r < 8; ++r) {
      const float f = rscale[wm * 16 + cr + r];
      o0[r] *= f;
      o1[r] *= f;
    }
#pragma unroll
    for (int k0 = 0; k0 < BC; k0 += 32) {
      v16h a  = load_a_frag(&Pb[(wm * 16) * LP + k0], LP);
      v16h b0 = load_bt_frag(&Vt[(wn * 32) * LV + k0], LV);
      v16h b1 = load_bt_frag(&Vt[(wn * 32 + 16) * LV + k0], LV);
      o0 = WMMA16(a, b0, o0);
      o1 = WMMA16(a, b1, o1);
    }
  }
  __syncthreads();

  // normalize and store: O[b][n][h*HD + d] as f16 for the output projection
#pragma unroll
  for (int r = 0; r < 8; ++r) {
    const int row = wm * 16 + cr + r;
    const float inv = 1.0f / rsum[row];
    const int gn = qt * BR + row;
    const size_t base = ((size_t)b * NSEQ + gn) * INTER + (size_t)h * HD;
    O[base + wn * 32 + cc]      = (half_t)(o0[r] * inv);
    O[base + wn * 32 + 16 + cc] = (half_t)(o1[r] * inv);
  }
}

// ---------------------------------------------------------------------------
// launcher
// ---------------------------------------------------------------------------
extern "C" void kernel_launch(void* const* d_in, const int* in_sizes, int n_in,
                              void* d_out, int out_size, void* d_ws, size_t ws_size,
                              hipStream_t stream) {
  const float* features = (const float*)d_in[0];   // [4,2048,1024]
  const float* W_qkv    = (const float*)d_in[1];   // [1024,3072]
  const float* W_out    = (const float*)d_in[2];   // [1024,1024]
  float* out = (float*)d_out;                      // [4,2048,1024] f32

  const size_t MN = (size_t)BATCH * NSEQ;          // 8192
  char* ws = (char*)d_ws;
  half_t* featH  = (half_t*)ws; ws += MN * DIMSZ * sizeof(half_t);                // 16 MB
  half_t* wqkvT  = (half_t*)ws; ws += (size_t)DIMSZ * 3 * INTER * sizeof(half_t); // 6 MB  [3N][K]
  half_t* woutT  = (half_t*)ws; ws += (size_t)INTER * DIMSZ * sizeof(half_t);     // 2 MB  [N][K]
  half_t* Qb     = (half_t*)ws; ws += MN * INTER * sizeof(half_t);                // 16 MB
  half_t* Kb     = (half_t*)ws; ws += MN * INTER * sizeof(half_t);                // 16 MB
  half_t* Vb     = (half_t*)ws; ws += MN * INTER * sizeof(half_t);                // 16 MB
  half_t* attH   = (half_t*)ws; ws += MN * INTER * sizeof(half_t);                // 16 MB

  // 1) precision conversion (+ one-time weight transposes to K-contiguous)
  f32_to_f16<<<4096, 256, 0, stream>>>(features, featH, (int)(MN * DIMSZ));
  f32_to_f16_T<<<dim3(3 * INTER / 32, DIMSZ / 32), 256, 0, stream>>>(
      W_qkv, wqkvT, DIMSZ, 3 * INTER);
  f32_to_f16_T<<<dim3(DIMSZ / 32, INTER / 32), 256, 0, stream>>>(
      W_out, woutT, INTER, DIMSZ);

  // 2) fused QKV projection, scatter into Q/K [B][H][N][HD], V [B][H][HD][N]
  gemm_f16<1><<<dim3(3 * INTER / BN, MN / BM), 256, 0, stream>>>(
      featH, wqkvT, nullptr, Qb, Kb, Vb, (int)MN, 3 * INTER, DIMSZ);

  // 3) flash attention (scores never touch HBM)
  flash_attn<<<dim3(BATCH * HEADS, NSEQ / BR), 256, 0, stream>>>(Qb, Kb, Vb, attH);

  // 4) output projection -> f32
  gemm_f16<0><<<dim3(DIMSZ / BN, MN / BM), 256, 0, stream>>>(
      attH, woutT, out, nullptr, nullptr, nullptr, (int)MN, DIMSZ, INTER);
}
